// RNade_75977971466989
// MI455X (gfx1250) — matrix operand order repaired and checked by reading, not compiled, not run on hardware
//
#include <hip/hip_runtime.h>

typedef __attribute__((ext_vector_type(16))) _Float16 v16h;
typedef __attribute__((ext_vector_type(8)))  _Float16 v8h;
typedef __attribute__((ext_vector_type(8)))  float    v8f;

// Problem dimensions (fixed by the reference)
constexpr int kB   = 256;
constexpr int kL   = 500;
constexpr int kD   = 3;
constexpr int kC   = 32;
constexpr int kH   = 512;
constexpr int kM   = 5;
constexpr int kI   = kD + kC;        // 35  per-step input dim
constexpr int kO   = kD * kM;        // 15
constexpr int kOut = 2 * kO + kM;    // 35  output cols per (b,l)

// Tiling
constexpr int kG  = 50;              // chunks over L
constexpr int kT  = kL / kG;         // 10 steps per chunk
constexpr int kBT = 32;              // batch tile rows per workgroup

// LDS strides (halfwords). All multiples of 8 halfs (16B) for ds_load_b128.
constexpr int XSTR = 72;             // x tile rows (K padded 35->64, +8 pad)
constexpr int WSTR = 72;             // W^T rows (512 rows x K=64)
constexpr int HSTR = 520;            // h rows (K=512, +8 pad)
constexpr int VSTR = 520;            // V^T rows (48 rows x K=512)
constexpr int NV   = 48;             // padded output cols (3 N-tiles)

// ---- WMMA fragment loaders (CDNA5 16x16x32 f16 layouts, ISA 7.12.2) ----
// A (16x32, row-major LDS tile): lane<16 -> K in [0,8)u[16,24); lane>=16 -> [8,16)u[24,32)
__device__ __forceinline__ v16h frag_a(const _Float16* buf, int stride,
                                       int m0, int k0, int ln, int hi) {
  const _Float16* p = buf + (m0 + ln) * stride + k0 + hi * 8;
  v8h lo = *(const v8h*)p;
  v8h hh = *(const v8h*)(p + 16);
  return __builtin_shufflevector(lo, hh, 0,1,2,3,4,5,6,7,8,9,10,11,12,13,14,15);
}
// B (32x16, stored transposed: buf[n][k]): lane<16 -> K in [0,8)u[16,24); lane>=16 -> [8,16)u[24,32)
__device__ __forceinline__ v16h frag_b(const _Float16* buf, int stride,
                                       int n0, int k0, int ln, int hi) {
  const _Float16* p = buf + (n0 + ln) * stride + k0 + hi * 8;
  v8h lo = *(const v8h*)p;
  v8h hh = *(const v8h*)(p + 16);
  return __builtin_shufflevector(lo, hh, 0,1,2,3,4,5,6,7,8,9,10,11,12,13,14,15);
}

// =====================================================================
// Phase A: per-chunk sums  C_g[b,h] = sum_{l in chunk g} (x_l @ W_enc[l])
// grid = (B/kBT, kG), block = 256 (8 wave32)
// =====================================================================
__global__ __launch_bounds__(256) void rnade_chunk_sums(
    const float* __restrict__ inputs, const float* __restrict__ z,
    const float* __restrict__ W_enc, float* __restrict__ csum) {
  __shared__ _Float16 Wt[kH * WSTR];   // W^T: [h][i], K padded to 64
  __shared__ _Float16 Xb[kBT * XSTR];  // x tile: [b_local][i]

  const int tid  = threadIdx.x;
  const int lane = tid & 31;
  const int wave = tid >> 5;
  const int hi   = lane >> 4;
  const int ln   = lane & 15;
  const int b0   = blockIdx.x * kBT;
  const int g    = blockIdx.y;

  for (int i = tid; i < kH * WSTR; i += 256) Wt[i] = (_Float16)0.f;
  for (int i = tid; i < kBT * XSTR; i += 256) Xb[i] = (_Float16)0.f;

  const int mt    = wave & 1;          // M-tile (2 x 16 rows)
  const int nbase = (wave >> 1) * 8;   // 8 N-tiles per wave (32 total)

  v8f acc[8];
  for (int j = 0; j < 8; ++j)
    acc[j] = (v8f){0.f,0.f,0.f,0.f,0.f,0.f,0.f,0.f};

  for (int s = 0; s < kT; ++s) {
    const int l = g * kT + s;
    __syncthreads();
    // stage x_l (concat inputs|z) as f16
    for (int idx = tid; idx < kBT * kI; idx += 256) {
      int bl = idx / kI, i = idx % kI;
      int b = b0 + bl;
      float v = (i < kD) ? inputs[(size_t)b * kL * kD + (size_t)l * kD + i]
                         : z[(size_t)b * kL * kC + (size_t)l * kC + (i - kD)];
      Xb[bl * XSTR + i] = (_Float16)v;
    }
    // stage W_enc[l]^T as f16 (coalesced global reads along h)
    const float* Wl = W_enc + (size_t)l * kI * kH;
    for (int idx = tid; idx < kI * kH; idx += 256) {
      int i = idx / kH, h = idx % kH;
      Wt[h * WSTR + i] = (_Float16)Wl[(size_t)i * kH + h];
    }
    __syncthreads();
    // GEMM1: [32,64] x [64,512] accumulate
    v16h a0 = frag_a(Xb, XSTR, mt * 16, 0,  ln, hi);
    v16h a1 = frag_a(Xb, XSTR, mt * 16, 32, ln, hi);
    for (int j = 0; j < 8; ++j) {
      int n0 = (nbase + j) * 16;
      v16h bw0 = frag_b(Wt, WSTR, n0, 0,  ln, hi);
      v16h bw1 = frag_b(Wt, WSTR, n0, 32, ln, hi);
      acc[j] = __builtin_amdgcn_wmma_f32_16x16x32_f16(false, a0, false, bw0,
                                                      (short)0, acc[j], false, false);
      acc[j] = __builtin_amdgcn_wmma_f32_16x16x32_f16(false, a1, false, bw1,
                                                      (short)0, acc[j], false, false);
    }
  }
  // write chunk sums
  for (int j = 0; j < 8; ++j) {
    int n0 = (nbase + j) * 16;
    for (int r = 0; r < 8; ++r) {
      int b = b0 + mt * 16 + r + hi * 8;
      int h = n0 + ln;
      csum[(size_t)g * kB * kH + (size_t)b * kH + h] = acc[j][r];
    }
  }
}

// =====================================================================
// Phase B: exclusive prefix over chunks, in place. One thread per (b,h).
// =====================================================================
__global__ void rnade_scan(float* __restrict__ csum) {
  int t = blockIdx.x * blockDim.x + threadIdx.x;
  if (t >= kB * kH) return;
  float run = 0.f;
  float* p = csum + t;
  for (int g = 0; g < kG; ++g) {
    float v = p[(size_t)g * kB * kH];
    p[(size_t)g * kB * kH] = run;
    run += v;
  }
}

// =====================================================================
// Phase C: per chunk, per batch tile: stream steps; h=relu(prefix acc),
// GEMM2 (h @ [V_mu|V_sigma|V_pi]) -> out, then GEMM1 updates prefix acc.
// =====================================================================
__global__ __launch_bounds__(256) void rnade_main(
    const float* __restrict__ inputs, const float* __restrict__ z,
    const float* __restrict__ W_enc, const float* __restrict__ b_enc,
    const float* __restrict__ V_mu, const float* __restrict__ b_mu,
    const float* __restrict__ V_sigma, const float* __restrict__ b_sigma,
    const float* __restrict__ V_pi, const float* __restrict__ b_pi,
    const float* __restrict__ P, float* __restrict__ out) {
  __shared__ _Float16 Wt[kH * WSTR];    // 73.7 KB
  __shared__ _Float16 Vt[NV * VSTR];    // 49.9 KB  V^T: [o][h]
  __shared__ _Float16 Hb[kBT * HSTR];   // 33.3 KB  h:   [b_local][h]
  __shared__ _Float16 Xb[kBT * XSTR];   //  4.6 KB

  const int tid  = threadIdx.x;
  const int lane = tid & 31;
  const int wave = tid >> 5;
  const int hi   = lane >> 4;
  const int ln   = lane & 15;
  const int b0   = blockIdx.x * kBT;
  const int g    = blockIdx.y;

  for (int i = tid; i < kH * WSTR; i += 256) Wt[i] = (_Float16)0.f;
  for (int i = tid; i < NV * VSTR; i += 256) Vt[i] = (_Float16)0.f;
  for (int i = tid; i < kBT * XSTR; i += 256) Xb[i] = (_Float16)0.f;

  const int mt    = wave & 1;
  const int nbase = (wave >> 1) * 8;

  // init prefix accumulator: a = P_g + b_enc  (exclusive prefix at l = g*kT)
  v8f acc[8];
  for (int j = 0; j < 8; ++j) {
    int n0 = (nbase + j) * 16;
    int h  = n0 + ln;
    float be = b_enc[h];
    for (int r = 0; r < 8; ++r) {
      int b = b0 + mt * 16 + r + hi * 8;
      acc[j][r] = P[(size_t)g * kB * kH + (size_t)b * kH + h] + be;
    }
  }

  for (int s = 0; s < kT; ++s) {
    const int l = g * kT + s;
    __syncthreads();  // previous step's LDS consumers are done

    // h = relu(acc)  (exclusive: before this step's GEMM1 update)
    for (int j = 0; j < 8; ++j) {
      int n0 = (nbase + j) * 16;
      for (int r = 0; r < 8; ++r)
        Hb[(mt * 16 + r + hi * 8) * HSTR + n0 + ln] = (_Float16)fmaxf(acc[j][r], 0.f);
    }
    // stage x_l
    for (int idx = tid; idx < kBT * kI; idx += 256) {
      int bl = idx / kI, i = idx % kI;
      int b = b0 + bl;
      float v = (i < kD) ? inputs[(size_t)b * kL * kD + (size_t)l * kD + i]
                         : z[(size_t)b * kL * kC + (size_t)l * kC + (i - kD)];
      Xb[bl * XSTR + i] = (_Float16)v;
    }
    // stage W_enc[l]^T
    const float* Wl = W_enc + (size_t)l * kI * kH;
    for (int idx = tid; idx < kI * kH; idx += 256) {
      int i = idx / kH, h = idx % kH;
      Wt[h * WSTR + i] = (_Float16)Wl[(size_t)i * kH + h];
    }
    // stage V^T = [V_mu|V_sigma|V_pi]^T (h-major index -> contiguous LDS writes)
    const float* Vmu = V_mu    + (size_t)l * kH * kO;
    const float* Vsg = V_sigma + (size_t)l * kH * kO;
    const float* Vpi = V_pi    + (size_t)l * kH * kM;
    for (int idx = tid; idx < kOut * kH; idx += 256) {
      int o = idx / kH, h = idx % kH;
      float v = (o < kO)     ? Vmu[(size_t)h * kO + o]
              : (o < 2*kO)   ? Vsg[(size_t)h * kO + (o - kO)]
                             : Vpi[(size_t)h * kM + (o - 2*kO)];
      Vt[o * VSTR + h] = (_Float16)v;
    }
    __syncthreads();

    // GEMM2: out[l] = h @ Vcat  (6 tiles on waves 0..5, full K=512)
    if (wave < 6) {
      const int mo = wave & 1;      // M-tile
      const int no = wave >> 1;     // N-tile (0..2)
      v8f oacc = (v8f){0.f,0.f,0.f,0.f,0.f,0.f,0.f,0.f};
      for (int ks = 0; ks < 16; ++ks) {
        v16h ah = frag_a(Hb, HSTR, mo * 16, ks * 32, ln, hi);
        v16h bv = frag_b(Vt, VSTR, no * 16, ks * 32, ln, hi);
        oacc = __builtin_amdgcn_wmma_f32_16x16x32_f16(false, ah, false, bv,
                                                      (short)0, oacc, false, false);
      }
      int col = no * 16 + ln;
      if (col < kOut) {
        float bias = (col < kO)   ? b_mu[(size_t)l * kO + col]
                   : (col < 2*kO) ? b_sigma[(size_t)l * kO + (col - kO)]
                                  : b_pi[(size_t)l * kM + (col - 2*kO)];
        for (int r = 0; r < 8; ++r) {
          int b = b0 + mo * 16 + r + hi * 8;
          out[(size_t)b * kL * kOut + (size_t)l * kOut + col] = oacc[r] + bias;
        }
      }
    }

    // GEMM1: acc += x_l @ W_enc[l]
    v16h a0 = frag_a(Xb, XSTR, mt * 16, 0,  ln, hi);
    v16h a1 = frag_a(Xb, XSTR, mt * 16, 32, ln, hi);
    for (int j = 0; j < 8; ++j) {
      int n0 = (nbase + j) * 16;
      v16h bw0 = frag_b(Wt, WSTR, n0, 0,  ln, hi);
      v16h bw1 = frag_b(Wt, WSTR, n0, 32, ln, hi);
      acc[j] = __builtin_amdgcn_wmma_f32_16x16x32_f16(false, a0, false, bw0,
                                                      (short)0, acc[j], false, false);
      acc[j] = __builtin_amdgcn_wmma_f32_16x16x32_f16(false, a1, false, bw1,
                                                      (short)0, acc[j], false, false);
    }
  }
}

extern "C" void kernel_launch(void* const* d_in, const int* in_sizes, int n_in,
                              void* d_out, int out_size, void* d_ws, size_t ws_size,
                              hipStream_t stream) {
  (void)in_sizes; (void)n_in; (void)out_size; (void)ws_size;
  const float* inputs  = (const float*)d_in[0];
  const float* z       = (const float*)d_in[1];
  const float* W_enc   = (const float*)d_in[2];
  const float* b_enc   = (const float*)d_in[3];
  const float* V_mu    = (const float*)d_in[4];
  const float* b_mu    = (const float*)d_in[5];
  const float* V_sigma = (const float*)d_in[6];
  const float* b_sigma = (const float*)d_in[7];
  const float* V_pi    = (const float*)d_in[8];
  const float* b_pi    = (const float*)d_in[9];
  float* out  = (float*)d_out;
  float* csum = (float*)d_ws;   // kG*kB*kH floats = 26.2 MB scratch

  dim3 grid(kB / kBT, kG);
  rnade_chunk_sums<<<grid, 256, 0, stream>>>(inputs, z, W_enc, csum);
  rnade_scan<<<dim3((kB * kH + 255) / 256), 256, 0, stream>>>(csum);
  rnade_main<<<grid, 256, 0, stream>>>(inputs, z, W_enc, b_enc,
                                       V_mu, b_mu, V_sigma, b_sigma,
                                       V_pi, b_pi, csum, out);
}
